// MixtralModel_26379689132542
// MI455X (gfx1250) — compile-verified, matrix operand order checked by reading, not run on hardware
//
#include <hip/hip_runtime.h>
#include <math.h>

#define T_ 2048
#define H_ 1024
#define HQ_ 16
#define HK_ 4
#define D_ 64
#define E_ 8
#define I_ 2048
#define TP_ (T_ + 32)   // padded key length for vT

typedef __attribute__((ext_vector_type(16))) __bf16 v16bf;
typedef __attribute__((ext_vector_type(8)))  float  v8f;
typedef __attribute__((ext_vector_type(4)))  int    v4i_a;

#if __has_builtin(__builtin_amdgcn_global_load_async_to_lds_b128) && __has_builtin(__builtin_amdgcn_s_wait_asynccnt)
#define HAVE_ASYNC 1
#else
#define HAVE_ASYNC 0
#endif

// ---------- small helpers ----------
__device__ __forceinline__ unsigned short f2bf(float f) {
    unsigned u = __float_as_uint(f);
    u += 0x7fffu + ((u >> 16) & 1u);          // round-to-nearest-even
    return (unsigned short)(u >> 16);
}
__device__ __forceinline__ float bf2f(unsigned short h) {
    return __uint_as_float(((unsigned)h) << 16);
}

// async 16B copy global -> LDS (falls back to sync copy if builtin missing)
__device__ __forceinline__ void async_cp16(const unsigned short* g, unsigned short* l) {
#if HAVE_ASYNC
    __builtin_amdgcn_global_load_async_to_lds_b128(
        (__attribute__((address_space(1))) v4i_a*)g,
        (__attribute__((address_space(3))) v4i_a*)l, 0, 0);
#else
    *(uint4*)l = *(const uint4*)g;
#endif
}
__device__ __forceinline__ void wait_async() {
#if HAVE_ASYNC
    __builtin_amdgcn_s_wait_asynccnt(0);
#endif
}

// Load one 16x32 bf16 A/B fragment (wave32 layout).
// p must already point at base + row*ld + k0 + hi*8  (16B aligned).
__device__ __forceinline__ v16bf load_frag16(const unsigned short* p) {
    union { uint4 u[2]; v16bf v; } t;
    t.u[0] = *(const uint4*)(p);
    t.u[1] = *(const uint4*)(p + 16);
    return t.v;
}

__device__ __forceinline__ v8f wmma_bf16(v16bf a, v16bf b, v8f c) {
    return __builtin_amdgcn_wmma_f32_16x16x32_bf16(false, a, false, b, (short)0, c, false, false);
}

// ---------- fp32 -> bf16 convert ----------
__global__ void k_convert(const float* __restrict__ src, unsigned short* __restrict__ dst, size_t n) {
    size_t i = (size_t)blockIdx.x * blockDim.x + threadIdx.x;
    size_t stride = (size_t)gridDim.x * blockDim.x;
    for (; i < n; i += stride) dst[i] = f2bf(src[i]);
}

// ---------- init: zero expert counters + vT pad region ----------
__global__ void k_init(int* __restrict__ cnt, unsigned short* __restrict__ vT) {
    int i = blockIdx.x * blockDim.x + threadIdx.x;
    if (i < E_) cnt[i] = 0;
    for (int j = i; j < HK_ * D_ * 32; j += gridDim.x * blockDim.x) {
        int hk = j / (D_ * 32);
        int d  = (j / 32) % D_;
        int tl = j % 32;
        vT[(size_t)hk * D_ * TP_ + (size_t)d * TP_ + T_ + tl] = 0;
    }
}

// ---------- RMSNorm (fp32 in) -> bf16 out ----------
__global__ __launch_bounds__(256) void k_rmsnorm_bf(const float* __restrict__ x,
                                                    const float* __restrict__ w,
                                                    unsigned short* __restrict__ out) {
    __shared__ float red[256];
    int t = blockIdx.x;
    const float* xr = x + (size_t)t * H_;
    float ss = 0.f;
    for (int i = threadIdx.x; i < H_; i += 256) { float v = xr[i]; ss += v * v; }
    red[threadIdx.x] = ss; __syncthreads();
    for (int s = 128; s > 0; s >>= 1) { if (threadIdx.x < s) red[threadIdx.x] += red[threadIdx.x + s]; __syncthreads(); }
    float rs = rsqrtf(red[0] / (float)H_ + 1e-5f);
    for (int i = threadIdx.x; i < H_; i += 256)
        out[(size_t)t * H_ + i] = f2bf(xr[i] * rs * w[i]);
}

// ---------- WMMA GEMM: C(MxN) = A(MxK,bf16) . W(NxK,bf16)^T  [+ add] ----------
// Block: 8 waves, tile M=32 x N=256. A tile double-buffered in LDS via async copy.
__global__ __launch_bounds__(256) void k_gemm_bf16(const unsigned short* __restrict__ A,
                                                   const unsigned short* __restrict__ W,
                                                   const float* __restrict__ add,
                                                   float* __restrict__ C,
                                                   int M, int N, int K) {
    __shared__ __align__(16) unsigned short sA[2][32 * 32];
    const int tid = threadIdx.x;
    const int wv = tid >> 5;
    const int lane = tid & 31;
    const int lr = lane & 15, hi = lane >> 4;
    const int m0 = blockIdx.x * 32;
    const int n0 = blockIdx.y * 256 + wv * 32;

    v8f acc[2][2];
#pragma unroll
    for (int mt = 0; mt < 2; ++mt)
#pragma unroll
        for (int nt = 0; nt < 2; ++nt)
#pragma unroll
            for (int r = 0; r < 8; ++r) acc[mt][nt][r] = 0.f;

    // chunk layout for A staging: 32 rows x 4 chunks of 8 bf16
    const int cr = tid >> 2, cq = tid & 3;      // valid for tid < 128
    // preload k-step 0
    if (tid < 128)
        async_cp16(A + (size_t)(m0 + cr) * K + cq * 8, &sA[0][cr * 32 + cq * 8]);
    wait_async();
    __syncthreads();

    const int ksteps = K / 32;
    for (int ks = 0; ks < ksteps; ++ks) {
        const int kc = ks & 1;
        if (ks + 1 < ksteps && tid < 128)
            async_cp16(A + (size_t)(m0 + cr) * K + (ks + 1) * 32 + cq * 8,
                       &sA[kc ^ 1][cr * 32 + cq * 8]);
        v16bf a0 = load_frag16(&sA[kc][lr * 32 + hi * 8]);
        v16bf a1 = load_frag16(&sA[kc][(16 + lr) * 32 + hi * 8]);
#pragma unroll
        for (int nt = 0; nt < 2; ++nt) {
            const unsigned short* wrow = W + (size_t)(n0 + nt * 16 + lr) * K + ks * 32 + hi * 8;
            if (ks + 1 < ksteps) __builtin_prefetch(wrow + 32, 0, 1);
            v16bf b = load_frag16(wrow);
            acc[0][nt] = wmma_bf16(a0, b, acc[0][nt]);
            acc[1][nt] = wmma_bf16(a1, b, acc[1][nt]);
        }
        wait_async();
        __syncthreads();
    }
#pragma unroll
    for (int mt = 0; mt < 2; ++mt)
#pragma unroll
        for (int nt = 0; nt < 2; ++nt)
#pragma unroll
            for (int r = 0; r < 8; ++r) {
                int m = m0 + mt * 16 + r + 8 * hi;
                int n = n0 + nt * 16 + lr;
                float v = acc[mt][nt][r];
                if (add) v += add[(size_t)m * N + n];
                C[(size_t)m * N + n] = v;
            }
}

// ---------- RoPE (NeoX) + layout: q_bf[h][t][d], k_bf[hk][t][d], vT[hk][d][t] ----------
__global__ __launch_bounds__(32) void k_rope(const float* __restrict__ qkv,
                                             const int* __restrict__ positions,
                                             unsigned short* __restrict__ qb,
                                             unsigned short* __restrict__ kb,
                                             unsigned short* __restrict__ vT) {
    int t = blockIdx.x;
    int hidx = blockIdx.y;          // 0..23 : 16 q heads, 4 k heads, 4 v heads
    int d2 = threadIdx.x;           // 0..31
    float pos = (float)positions[t];
    if (hidx < HQ_ + HK_) {
        float invf = __powf(10000.f, -(float)d2 / 32.f);
        float ang = pos * invf, s, c;
        __sincosf(ang, &s, &c);
        const float* base;
        unsigned short* ob;
        if (hidx < HQ_) {
            base = qkv + (size_t)t * 1536 + hidx * D_;
            ob = qb + (size_t)hidx * T_ * D_ + (size_t)t * D_;
        } else {
            int kh = hidx - HQ_;
            base = qkv + (size_t)t * 1536 + HQ_ * D_ + kh * D_;
            ob = kb + (size_t)kh * T_ * D_ + (size_t)t * D_;
        }
        float x1 = base[d2], x2 = base[d2 + 32];
        ob[d2]      = f2bf(x1 * c - x2 * s);
        ob[d2 + 32] = f2bf(x2 * c + x1 * s);
    } else {
        int vh = hidx - HQ_ - HK_;
        const float* base = qkv + (size_t)t * 1536 + (HQ_ + HK_) * D_ + vh * D_;
        unsigned short* vb = vT + (size_t)vh * D_ * TP_;
        vb[(size_t)d2 * TP_ + t]        = f2bf(base[d2]);
        vb[(size_t)(d2 + 32) * TP_ + t] = f2bf(base[d2 + 32]);
    }
}

// ---------- flash attention: one wave per (16-query tile, head) ----------
__global__ __launch_bounds__(32) void k_attn(const unsigned short* __restrict__ qb,
                                             const unsigned short* __restrict__ kb,
                                             const unsigned short* __restrict__ vT,
                                             unsigned short* __restrict__ attn) {
    __shared__ __align__(16) unsigned short pl[16 * 32];
    const int lane = threadIdx.x & 31;
    const int lr = lane & 15, hi = lane >> 4;
    const int qt = blockIdx.x;
    const int h = blockIdx.y;
    const int hk = h / (HQ_ / HK_);
    for (int i = lane; i < 16 * 32; i += 32) pl[i] = 0;
    __syncthreads();

    const unsigned short* qrow = qb + (size_t)h * T_ * D_ + (size_t)(qt * 16 + lr) * D_ + hi * 8;
    v16bf qa0 = load_frag16(qrow);
    v16bf qa1 = load_frag16(qrow + 32);
    const unsigned short* kbase = kb + (size_t)hk * T_ * D_;
    const unsigned short* vbase = vT + (size_t)hk * D_ * TP_;

    float Mr[8], Lr[8];
#pragma unroll
    for (int r = 0; r < 8; ++r) { Mr[r] = -INFINITY; Lr[r] = 0.f; }
    v8f acc[4];
#pragma unroll
    for (int nt = 0; nt < 4; ++nt)
#pragma unroll
        for (int r = 0; r < 8; ++r) acc[nt][r] = 0.f;

    for (int j = 0; j <= qt; ++j) {
        v8f s;
#pragma unroll
        for (int r = 0; r < 8; ++r) s[r] = 0.f;
        const unsigned short* krow = kbase + (size_t)(j * 16 + lr) * D_ + hi * 8;
        s = wmma_bf16(qa0, load_frag16(krow), s);
        s = wmma_bf16(qa1, load_frag16(krow + 32), s);
        float p8[8], sc[8];
#pragma unroll
        for (int r = 0; r < 8; ++r) {
            float v = s[r] * 0.125f;    // D^-0.5
            if (j == qt) {
                int kg = j * 16 + lr, qg = qt * 16 + r + 8 * hi;
                if (kg > qg) v = -INFINITY;
            }
            float mx = v;
            mx = fmaxf(mx, __shfl_xor(mx, 1, 16));
            mx = fmaxf(mx, __shfl_xor(mx, 2, 16));
            mx = fmaxf(mx, __shfl_xor(mx, 4, 16));
            mx = fmaxf(mx, __shfl_xor(mx, 8, 16));
            float Mn = fmaxf(Mr[r], mx);
            sc[r] = __expf(Mr[r] - Mn);
            float p = __expf(v - Mn);
            float ps = p;
            ps += __shfl_xor(ps, 1, 16);
            ps += __shfl_xor(ps, 2, 16);
            ps += __shfl_xor(ps, 4, 16);
            ps += __shfl_xor(ps, 8, 16);
            Lr[r] = Lr[r] * sc[r] + ps;
            Mr[r] = Mn;
            p8[r] = p;
        }
#pragma unroll
        for (int nt = 0; nt < 4; ++nt)
#pragma unroll
            for (int r = 0; r < 8; ++r) acc[nt][r] *= sc[r];
#pragma unroll
        for (int r = 0; r < 8; ++r) pl[(r + 8 * hi) * 32 + lr] = f2bf(p8[r]);
        __syncthreads();
        v16bf pa = load_frag16(pl + lr * 32 + hi * 8);
#pragma unroll
        for (int nt = 0; nt < 4; ++nt) {
            const unsigned short* vrow = vbase + (size_t)(nt * 16 + lr) * TP_ + j * 16 + hi * 8;
            acc[nt] = wmma_bf16(pa, load_frag16(vrow), acc[nt]);
        }
        __syncthreads();
    }
#pragma unroll
    for (int r = 0; r < 8; ++r) {
        float inv = 1.f / Lr[r];
        int q = qt * 16 + r + 8 * hi;
#pragma unroll
        for (int nt = 0; nt < 4; ++nt)
            attn[(size_t)q * (HQ_ * D_) + h * D_ + nt * 16 + lr] = f2bf(acc[nt][r] * inv);
    }
}

// ---------- router ----------
__global__ __launch_bounds__(64) void k_route(const unsigned short* __restrict__ h2,
                                              const float* __restrict__ gate_w,
                                              int* __restrict__ cnt,
                                              int* __restrict__ toklist,
                                              float* __restrict__ wlist) {
    __shared__ float red[E_ * 64];
    int t = blockIdx.x, tid = threadIdx.x;
    float a[E_];
#pragma unroll
    for (int e = 0; e < E_; ++e) a[e] = 0.f;
    for (int k = tid; k < H_; k += 64) {
        float hv = bf2f(h2[(size_t)t * H_ + k]);
#pragma unroll
        for (int e = 0; e < E_; ++e) a[e] += hv * gate_w[e * H_ + k];
    }
#pragma unroll
    for (int e = 0; e < E_; ++e) red[e * 64 + tid] = a[e];
    __syncthreads();
    if (tid < E_) {
        float s = 0.f;
        for (int i = 0; i < 64; ++i) s += red[tid * 64 + i];
        red[tid] = s;
    }
    __syncthreads();
    if (tid == 0) {
        float lg[E_], mx = -INFINITY;
        for (int e = 0; e < E_; ++e) { lg[e] = red[e]; mx = fmaxf(mx, lg[e]); }
        float sum = 0.f;
        for (int e = 0; e < E_; ++e) { lg[e] = __expf(lg[e] - mx); sum += lg[e]; }
        for (int e = 0; e < E_; ++e) lg[e] /= sum;
        int i0 = 0; for (int e = 1; e < E_; ++e) if (lg[e] > lg[i0]) i0 = e;
        int i1 = (i0 == 0) ? 1 : 0;
        for (int e = 0; e < E_; ++e) if (e != i0 && lg[e] > lg[i1]) i1 = e;
        float w0 = lg[i0], w1 = lg[i1], ws = w0 + w1;
        w0 /= ws; w1 /= ws;
        int s0 = atomicAdd(&cnt[i0], 1);
        toklist[i0 * T_ + s0] = t;               // rank 0
        wlist[i0 * T_ + s0] = w0;
        int s1 = atomicAdd(&cnt[i1], 1);
        toklist[i1 * T_ + s1] = t | (1 << 16);   // rank 1
        wlist[i1 * T_ + s1] = w1;
    }
}

// ---------- MoE: 16-token tile per (expert, tile). 8 waves/block ----------
// Gathered 16x1024 activation tile staged once in LDS via async copy.
__global__ __launch_bounds__(256) void k_moe(const unsigned short* __restrict__ h2,
                                             const unsigned short* __restrict__ w1b,
                                             const unsigned short* __restrict__ w3b,
                                             const unsigned short* __restrict__ w2b,
                                             const int* __restrict__ cnt,
                                             const int* __restrict__ toklist,
                                             const float* __restrict__ wlist,
                                             float* __restrict__ part) {
    __shared__ __align__(16) unsigned short sA[16 * H_];      // 32 KB gathered activations
    __shared__ __align__(16) unsigned short sg[16 * 128];     // gated chunk
    __shared__ int s_tok[16], s_rank[16], s_valid[16];
    __shared__ float s_w[16];
    const int e = blockIdx.y;
    const int base = blockIdx.x * 16;
    const int cntE = cnt[e];
    if (base >= cntE) return;                       // uniform exit: EXEC stays full
    if (threadIdx.x < 16) {
        int slot = base + threadIdx.x;
        int use = (slot < cntE) ? slot : (cntE - 1);
        int enc = toklist[e * T_ + use];
        s_tok[threadIdx.x] = enc & 0xffff;
        s_rank[threadIdx.x] = enc >> 16;
        s_w[threadIdx.x] = wlist[e * T_ + use];
        s_valid[threadIdx.x] = (slot < cntE);
    }
    __syncthreads();

    const int wv = threadIdx.x >> 5;
    const int lane = threadIdx.x & 31;
    const int lr = lane & 15, hi = lane >> 4;

    // stage gathered A tile: 16 rows x 1024 cols = 2048 chunks of 8 bf16
    {
        for (int c = threadIdx.x; c < 16 * (H_ / 8); c += 256) {
            int row = c >> 7;           // 128 chunks per row
            int col8 = c & 127;
            async_cp16(h2 + (size_t)s_tok[row] * H_ + col8 * 8, &sA[row * H_ + col8 * 8]);
        }
        wait_async();
        __syncthreads();
    }

    const unsigned short* arow = sA + lr * H_ + hi * 8;
    const unsigned short* w1e = w1b + (size_t)e * I_ * H_;
    const unsigned short* w3e = w3b + (size_t)e * I_ * H_;
    const unsigned short* w2e = w2b + (size_t)e * H_ * I_;

    v8f oacc[8];
#pragma unroll
    for (int j = 0; j < 8; ++j)
#pragma unroll
        for (int r = 0; r < 8; ++r) oacc[j][r] = 0.f;

    for (int c = 0; c < I_ / 128; ++c) {
        const int i0 = c * 128;
        v8f a1, a3;
#pragma unroll
        for (int r = 0; r < 8; ++r) { a1[r] = 0.f; a3[r] = 0.f; }
        const unsigned short* r1 = w1e + (size_t)(i0 + wv * 16 + lr) * H_ + hi * 8;
        const unsigned short* r3 = w3e + (size_t)(i0 + wv * 16 + lr) * H_ + hi * 8;
        for (int k0 = 0; k0 < H_; k0 += 32) {
            v16bf a = load_frag16(arow + k0);
            a1 = wmma_bf16(a, load_frag16(r1 + k0), a1);
            a3 = wmma_bf16(a, load_frag16(r3 + k0), a3);
        }
        __syncthreads();   // previous chunk's phase-2 reads done
#pragma unroll
        for (int r = 0; r < 8; ++r) {
            float x1 = a1[r], x3 = a3[r];
            float g = (x1 / (1.f + __expf(-x1))) * x3;   // silu(h1)*h3
            sg[(r + 8 * hi) * 128 + wv * 16 + lr] = f2bf(g);
        }
        __syncthreads();
        for (int kl = 0; kl < 128; kl += 32) {
            v16bf pa = load_frag16(sg + lr * 128 + kl + hi * 8);
#pragma unroll
            for (int j = 0; j < 8; ++j) {
                const unsigned short* wr = w2e + (size_t)(wv * 128 + j * 16 + lr) * I_ + i0 + kl + hi * 8;
                oacc[j] = wmma_bf16(pa, load_frag16(wr), oacc[j]);
            }
        }
    }
#pragma unroll
    for (int r = 0; r < 8; ++r) {
        int m = r + 8 * hi;
        if (!s_valid[m]) continue;
        int tok = s_tok[m], rank = s_rank[m];
        float wgt = s_w[m];
        float* dst = part + (size_t)rank * T_ * H_ + (size_t)tok * H_;
#pragma unroll
        for (int j = 0; j < 8; ++j)
            dst[wv * 128 + j * 16 + lr] = oacc[j][r] * wgt;
    }
}

// ---------- final: out = rmsnorm(resid1 + part0 + part1, norm_next_w) ----------
__global__ __launch_bounds__(256) void k_final(const float* __restrict__ resid1,
                                               const float* __restrict__ part,
                                               const float* __restrict__ w,
                                               float* __restrict__ out) {
    __shared__ float red[256];
    int t = blockIdx.x;
    float y[4], ss = 0.f;
#pragma unroll
    for (int j = 0; j < 4; ++j) {
        int i = threadIdx.x + j * 256;
        float v = resid1[(size_t)t * H_ + i] + part[(size_t)t * H_ + i] + part[(size_t)T_ * H_ + (size_t)t * H_ + i];
        y[j] = v; ss += v * v;
    }
    red[threadIdx.x] = ss; __syncthreads();
    for (int s = 128; s > 0; s >>= 1) { if (threadIdx.x < s) red[threadIdx.x] += red[threadIdx.x + s]; __syncthreads(); }
    float rs = rsqrtf(red[0] / (float)H_ + 1e-5f);
#pragma unroll
    for (int j = 0; j < 4; ++j) {
        int i = threadIdx.x + j * 256;
        out[(size_t)t * H_ + i] = y[j] * rs * w[i];
    }
}

// ---------- launcher ----------
static inline size_t al256(size_t x) { return (x + 255) & ~(size_t)255; }

extern "C" void kernel_launch(void* const* d_in, const int* in_sizes, int n_in,
                              void* d_out, int out_size, void* d_ws, size_t ws_size,
                              hipStream_t stream) {
    const float* x        = (const float*)d_in[0];
    const int*   pos      = (const int*)d_in[1];
    const float* qkv_w    = (const float*)d_in[2];
    const float* o_w      = (const float*)d_in[3];
    const float* norm_in  = (const float*)d_in[4];
    const float* norm_post= (const float*)d_in[5];
    const float* norm_next= (const float*)d_in[6];
    const float* gate_w   = (const float*)d_in[7];
    const float* w1       = (const float*)d_in[8];
    const float* w3       = (const float*)d_in[9];
    const float* w2       = (const float*)d_in[10];
    float* out = (float*)d_out;

    char* ws = (char*)d_ws;
    size_t off = 0;
    auto alloc = [&](size_t bytes) { char* p = ws + off; off = al256(off + bytes); return p; };
    unsigned short* qkvw_bf = (unsigned short*)alloc((size_t)1536 * H_ * 2);
    unsigned short* ow_bf   = (unsigned short*)alloc((size_t)H_ * H_ * 2);
    unsigned short* w1_bf   = (unsigned short*)alloc((size_t)E_ * I_ * H_ * 2);
    unsigned short* w3_bf   = (unsigned short*)alloc((size_t)E_ * I_ * H_ * 2);
    unsigned short* w2_bf   = (unsigned short*)alloc((size_t)E_ * H_ * I_ * 2);
    unsigned short* h_bf    = (unsigned short*)alloc((size_t)T_ * H_ * 2);
    float*          qkv     = (float*)alloc((size_t)T_ * 1536 * 4);
    unsigned short* q_bf    = (unsigned short*)alloc((size_t)HQ_ * T_ * D_ * 2);
    unsigned short* k_bf    = (unsigned short*)alloc((size_t)HK_ * T_ * D_ * 2);
    unsigned short* vT_bf   = (unsigned short*)alloc((size_t)HK_ * D_ * TP_ * 2);
    unsigned short* attn_bf = (unsigned short*)alloc((size_t)T_ * H_ * 2);
    float*          resid1  = (float*)alloc((size_t)T_ * H_ * 4);
    unsigned short* h2_bf   = (unsigned short*)alloc((size_t)T_ * H_ * 2);
    int*            cnt     = (int*)alloc(E_ * 4);
    int*            toklist = (int*)alloc((size_t)E_ * T_ * 4);
    float*          wlist   = (float*)alloc((size_t)E_ * T_ * 4);
    float*          part    = (float*)alloc((size_t)2 * T_ * H_ * 4);
    (void)ws_size; (void)n_in; (void)in_sizes; (void)out_size;

    // 1) weight conversion to bf16
    k_convert<<<4096, 256, 0, stream>>>(qkv_w, qkvw_bf, (size_t)1536 * H_);
    k_convert<<<4096, 256, 0, stream>>>(o_w, ow_bf, (size_t)H_ * H_);
    k_convert<<<4096, 256, 0, stream>>>(w1, w1_bf, (size_t)E_ * I_ * H_);
    k_convert<<<4096, 256, 0, stream>>>(w3, w3_bf, (size_t)E_ * I_ * H_);
    k_convert<<<4096, 256, 0, stream>>>(w2, w2_bf, (size_t)E_ * H_ * I_);
    // 2) init counters + vT pad
    k_init<<<32, 256, 0, stream>>>(cnt, vT_bf);
    // 3) input rmsnorm -> bf16
    k_rmsnorm_bf<<<T_, 256, 0, stream>>>(x, norm_in, h_bf);
    // 4) qkv = h @ qkv_w^T
    k_gemm_bf16<<<dim3(T_ / 32, 1536 / 256), 256, 0, stream>>>(h_bf, qkvw_bf, nullptr, qkv, T_, 1536, H_);
    // 5) rope + kv layout
    k_rope<<<dim3(T_, HQ_ + 2 * HK_), 32, 0, stream>>>(qkv, pos, q_bf, k_bf, vT_bf);
    // 6) attention
    k_attn<<<dim3(T_ / 16, HQ_), 32, 0, stream>>>(q_bf, k_bf, vT_bf, attn_bf);
    // 7) resid1 = x + attn @ o_w^T
    k_gemm_bf16<<<dim3(T_ / 32, H_ / 256), 256, 0, stream>>>(attn_bf, ow_bf, x, resid1, T_, H_, H_);
    // 8) post-attn rmsnorm -> bf16
    k_rmsnorm_bf<<<T_, 256, 0, stream>>>(resid1, norm_post, h2_bf);
    // 9) routing
    k_route<<<T_, 64, 0, stream>>>(h2_bf, gate_w, cnt, toklist, wlist);
    // 10) MoE expert GEMMs (sparse, per-expert tiles)
    k_moe<<<dim3(T_ / 16, E_), 256, 0, stream>>>(h2_bf, w1_bf, w3_bf, w2_bf, cnt, toklist, wlist, part);
    // 11) final residual + rmsnorm
    k_final<<<T_, 256, 0, stream>>>(resid1, part, norm_next, out);
}
